// MultiHeadAttention_37623913513540
// MI455X (gfx1250) — matrix operand order repaired and checked by reading, not compiled, not run on hardware
//
#include <hip/hip_runtime.h>

// ---------------------------------------------------------------------------
// MLA latent attention for MI455X (gfx1250, wave32, WMMA + TDM).
// All GEMM-shaped work runs on v_wmma_f32_16x16x32_bf16 (fp32 accumulate).
// Attention KV tiles are staged LDS-side with the Tensor Data Mover.
// ---------------------------------------------------------------------------

typedef __bf16 bf16;
typedef __attribute__((ext_vector_type(16))) __bf16 v16bf;
typedef __attribute__((ext_vector_type(8)))  __bf16 v8bf;
typedef __attribute__((ext_vector_type(8)))  float  v8f;
typedef __attribute__((ext_vector_type(4)))  unsigned int v4u;
typedef __attribute__((ext_vector_type(8)))  int v8i;
typedef __attribute__((ext_vector_type(4)))  int v4i;

#define DEV __device__ __forceinline__

constexpr int Bdim = 2;
constexpr int Tdim = 2048;
constexpr int Cdim = 1024;
constexpr int Hdim = 16;
constexpr int Ldim = 288;   // 18 tiles of 16, 9 K-chunks of 32

DEV v8f vzero8() {
  v8f v;
#pragma unroll
  for (int i = 0; i < 8; ++i) v[i] = 0.f;
  return v;
}

DEV v8f wmma_bf16(v16bf a, v16bf b, v8f c) {
  return __builtin_amdgcn_wmma_f32_16x16x32_bf16(
      /*neg_a=*/false, a, /*neg_b=*/false, b,
      /*c_mod=*/(short)0, c, /*reuse_a=*/false, /*reuse_b=*/false);
}

// A-fragment (16x32 bf16), contiguous k order, same as B: lane = hi*16 + r;
// element e -> K = hi*16 + e. Two 16B LDS loads.
DEV v16bf lds_a_frag(const bf16* base, int stride, int r, int k0, int hi) {
  const bf16* p = base + r * stride + k0 + hi * 16;
  v8bf lo = *(const v8bf*)(p);
  v8bf hh = *(const v8bf*)(p + 8);
  v16bf o;
#pragma unroll
  for (int e = 0; e < 8; ++e) { o[e] = lo[e]; o[e + 8] = hh[e]; }
  return o;
}

// B-fragment (32x16 bf16) from an LDS tile stored TRANSPOSED (Bt[n][k]):
// lane = hi*16 + col; element e -> K = hi*16 + e. 32 contiguous bytes.
DEV v16bf lds_bt_frag(const bf16* base, int stride, int col, int k0, int hi) {
  const bf16* p = base + col * stride + k0 + hi * 16;
  v8bf lo = *(const v8bf*)(p);
  v8bf hh = *(const v8bf*)(p + 8);
  v16bf o;
#pragma unroll
  for (int e = 0; e < 8; ++e) { o[e] = lo[e]; o[e + 8] = hh[e]; }
  return o;
}

// B-fragment gathered from a ROW-MAJOR K x N LDS tile (strided).
DEV v16bf lds_b_rowmajor_frag(const bf16* base, int stride, int col, int hi) {
  v16bf o;
#pragma unroll
  for (int e = 0; e < 16; ++e) o[e] = base[(hi * 16 + e) * stride + col];
  return o;
}

// ---------------------------------------------------------------------------
// Tensor Data Mover: 2D bf16 tile (rows x Ldim, contiguous rows) -> LDS.
// D# built per CDNA5 ISA ch.10.8: group0 {count=1, lds_addr, global_addr,
// type=2}; group1 {data_size=1(2B), tensor/tile dims, dim0 stride}.
// ---------------------------------------------------------------------------
#if __has_builtin(__builtin_amdgcn_tensor_load_to_lds)
#define MLA_USE_TDM 1
#else
#define MLA_USE_TDM 0
#endif

DEV void tdm_load_rows(unsigned lds_byte_off, const bf16* src, unsigned rows) {
  const unsigned long long ga = (unsigned long long)(uintptr_t)src;
  const unsigned row_elems = (unsigned)Ldim;
  v4u g0;
  g0[0] = 1u;                                   // count=1, user descriptor
  g0[1] = lds_byte_off;                         // D#.lds_addr
  g0[2] = (unsigned)(ga & 0xffffffffu);         // global_addr[31:0]
  g0[3] = (unsigned)((ga >> 32) & 0x01ffffffu)  // global_addr[56:32]
        | (2u << 30);                           // type=2 ("image")
  v8i g1;
  g1[0] = 0x00010000;                           // data_size=1 (2B), no pad/mask
  g1[1] = (int)((row_elems & 0xffffu) << 16);   // tensor_dim0[15:0]
  g1[2] = (int)(((row_elems >> 16) & 0xffffu)   // tensor_dim0[31:16]
              | ((rows & 0xffffu) << 16));      // tensor_dim1[15:0]
  g1[3] = (int)(((rows >> 16) & 0xffffu)        // tensor_dim1[31:16]
              | ((row_elems & 0xffffu) << 16)); // tile_dim0 = row length
  g1[4] = (int)(rows & 0xffffu);                // tile_dim1 = rows, tile_dim2=0
  g1[5] = (int)row_elems;                       // tensor_dim0_stride[31:0]
  g1[6] = 0;                                    // stride msbs / dim1_stride lo
  g1[7] = 0;
  const v4i z4 = {0, 0, 0, 0};
#if MLA_USE_TDM
#if defined(__clang_major__) && __clang_major__ >= 23
  const v8i z8 = {0, 0, 0, 0, 0, 0, 0, 0};
  __builtin_amdgcn_tensor_load_to_lds(g0, g1, z4, z4, z8, 0);
#else
  __builtin_amdgcn_tensor_load_to_lds(g0, g1, z4, z4, 0);
#endif
#else
  (void)g0; (void)g1; (void)z4;
#endif
}

DEV void tdm_wait0() {
#if __has_builtin(__builtin_amdgcn_s_wait_tensorcnt)
  __builtin_amdgcn_s_wait_tensorcnt(0);
#else
  asm volatile("s_wait_tensorcnt 0x0" ::: "memory");
#endif
  asm volatile("" ::: "memory");
}

DEV unsigned lds_offset_of(const void* p) {
  // Flat LDS aperture: addr[31:0] is the LDS byte offset (ISA 10.2).
  return (unsigned)(unsigned long long)(uintptr_t)p;
}

// Manual fallback staging (contiguous rows*Ldim bf16, wave32).
DEV void stage_rows(bf16* dst, const bf16* src, int rows, int lane) {
  const int chunks = rows * (Ldim / 8);
  for (int chunk = lane; chunk < chunks; chunk += 32) {
    const int off = chunk * 8;
    *(v8bf*)(dst + off) = *(const v8bf*)(src + off);
  }
}

// ---------------------------------------------------------------------------
// f32 -> bf16 conversion
// ---------------------------------------------------------------------------
__global__ __launch_bounds__(256) void cvt_f32_bf16(const float* __restrict__ in,
                                                    bf16* __restrict__ out, int n) {
  int i = blockIdx.x * blockDim.x + threadIdx.x;
  if (i < n) out[i] = (bf16)in[i];
}

// ---------------------------------------------------------------------------
// WMMA GEMM:  Out[M,N] = A[M,K] * B[K,N] + bias[N]
// MODE 0: plain row-major, Out bf16                       (Wdkv)
// MODE 1: B = Wd_w[head] (grid.z), Out -> [B,H,T,L] bf16  (q)
// MODE 2: A gathered from y[B,H,T,L], Out f32             (projection)
// ---------------------------------------------------------------------------
template <int MODE>
__global__ __launch_bounds__(256) void gemm_wmma_bf16(
    const bf16* __restrict__ A, const bf16* __restrict__ Bm,
    const float* __restrict__ bias, void* __restrict__ Out,
    int M, int N, int K) {
  __shared__ bf16 As[64 * 40];     // A tile [64][32], stride 40
  __shared__ bf16 Bts[128 * 40];   // B tile transposed [n:128][k:32]

  const int tid  = threadIdx.x;
  const int lane = tid & 31;
  const int wave = tid >> 5;
  const int wm   = wave & 3;
  const int wn   = wave >> 2;
  const int hi   = lane >> 4;
  const int c    = lane & 15;
  const int m0   = blockIdx.y * 64;
  const int n0   = blockIdx.x * 128;
  const int z    = blockIdx.z;

  const bf16* Bp = Bm + (size_t)z * K * N;

  v8f acc[4];
#pragma unroll
  for (int t = 0; t < 4; ++t) acc[t] = vzero8();

  const int ar = tid >> 2;          // 0..63
  const int ac = (tid & 3) << 3;    // 0,8,16,24

  for (int k0 = 0; k0 < K; k0 += 32) {
    // ---- stage A tile: one global_load_b128 per thread ----
    {
      const int m = m0 + ar;
      const int k = k0 + ac;
      size_t idx;
      if constexpr (MODE == 2) {    // gather from y[B,H,T,L]: k = h*L + l
        const int hh = k / Ldim;
        const int l  = k - hh * Ldim;
        const int bb = m >> 11;
        const int tt = m & (Tdim - 1);
        idx = ((size_t)(bb * Hdim + hh) * Tdim + tt) * Ldim + l;
      } else {
        idx = (size_t)m * K + k;
      }
      *(v8bf*)(As + ar * 40 + ac) = *(const v8bf*)(A + idx);
    }
    // ---- stage B tile: vector global load, transposed scalar LDS write ----
#pragma unroll
    for (int t = 0; t < 2; ++t) {
      const int chunk = tid * 2 + t;          // 0..511 covers 32k x 128n
      const int kk = chunk >> 4;
      const int nn = (chunk & 15) << 3;
      const int n = n0 + nn;
      if (n + 7 < N) {
        const v8bf v = *(const v8bf*)(Bp + (size_t)(k0 + kk) * N + n);
#pragma unroll
        for (int j = 0; j < 8; ++j) Bts[(nn + j) * 40 + kk] = v[j];
      } else {
#pragma unroll
        for (int j = 0; j < 8; ++j)
          Bts[(nn + j) * 40 + kk] =
              (n + j < N) ? Bp[(size_t)(k0 + kk) * N + n + j] : (bf16)0.f;
      }
    }
    __syncthreads();

    const v16bf aF = lds_a_frag(As + wm * 16 * 40, 40, c, 0, hi);
#pragma unroll
    for (int t = 0; t < 4; ++t) {
      const v16bf bF = lds_bt_frag(Bts + (wn * 64 + t * 16) * 40, 40, c, 0, hi);
      acc[t] = wmma_bf16(aF, bF, acc[t]);
    }
    __syncthreads();
  }

  // ---- epilogue: C layout vgpr i -> row hi*8+i, col = lane&15 ----
#pragma unroll
  for (int t = 0; t < 4; ++t) {
    const int col = n0 + wn * 64 + t * 16 + c;
    if (col >= N) continue;
    const float bv = bias[(MODE == 1 ? z * N : 0) + col];
#pragma unroll
    for (int i = 0; i < 8; ++i) {
      const int row = m0 + wm * 16 + hi * 8 + i;
      const float v = acc[t][i] + bv;
      if constexpr (MODE == 2) {
        ((float*)Out)[(size_t)row * N + col] = v;
      } else if constexpr (MODE == 1) {
        const int bb = row >> 11;
        const int tt = row & (Tdim - 1);
        ((bf16*)Out)[((size_t)(bb * Hdim + z) * Tdim + tt) * Ldim + col] = (bf16)v;
      } else {
        ((bf16*)Out)[(size_t)row * N + col] = (bf16)v;
      }
    }
  }
}

// ---------------------------------------------------------------------------
// Flash latent attention. One wave per (b, h, 16-query tile).
// KV blocks staged by the Tensor Data Mover (TENSORcnt), q tile likewise.
//   S[16,32] = q(16x288) * kv(32x288)^T    (9 K-chunks x 2 N-tiles)
//   online causal softmax; Y[16,288] += P(16x32) * kv(32x288) (18 N-tiles)
// ---------------------------------------------------------------------------
__global__ __launch_bounds__(32) void mla_flash_attn(
    const bf16* __restrict__ q, const bf16* __restrict__ kv,
    bf16* __restrict__ y) {
  __shared__ bf16 qs[16 * Ldim];   // q tile, row-major (A operand)
  __shared__ bf16 kvs[32 * Ldim];  // kv block row-major (Bt for scores, B for Y)
  __shared__ bf16 ps[16 * 40];     // P tile (A operand for Y)

  const int lane = threadIdx.x;
  const int hi = lane >> 4, c = lane & 15;
  const int bid = blockIdx.x;
  const int qt = bid & 127;
  const int h  = (bid >> 7) & (Hdim - 1);
  const int b  = bid >> 11;

  const size_t qbase = ((size_t)(b * Hdim + h) * Tdim + qt * 16) * Ldim;
#if MLA_USE_TDM
  const unsigned qs_off  = lds_offset_of(qs);
  const unsigned kvs_off = lds_offset_of(kvs);
  tdm_load_rows(qs_off, q + qbase, 16);
#else
  stage_rows(qs, q + qbase, 16, lane);
#endif

  v8f acc[18];
#pragma unroll
  for (int t = 0; t < 18; ++t) acc[t] = vzero8();
  float mrow[8], lrow[8];
#pragma unroll
  for (int i = 0; i < 8; ++i) { mrow[i] = -1e30f; lrow[i] = 0.f; }

  const float sc = 0.125f;                    // 1/sqrt(d_head=64)
  const int kbmax = (qt * 16 + 15) >> 5;

  for (int kb = 0; kb <= kbmax; ++kb) {
    const size_t kbase = ((size_t)b * Tdim + kb * 32) * Ldim;
    // prior DS reads of kvs must retire before the DMA overwrites it
    asm volatile("s_wait_dscnt 0x0" ::: "memory");
#if MLA_USE_TDM
    tdm_load_rows(kvs_off, kv + kbase, 32);
    tdm_wait0();                               // also covers the q tile on kb=0
#else
    stage_rows(kvs, kv + kbase, 32, lane);
    asm volatile("s_wait_dscnt 0x0" ::: "memory");
#endif

    // ---- scores: two 16x16 tiles over 32 keys ----
    v8f s0 = vzero8(), s1 = vzero8();
#pragma unroll
    for (int kc = 0; kc < 9; ++kc) {
      const v16bf aF = lds_a_frag(qs, Ldim, c, kc * 32, hi);
      const v16bf b0 = lds_bt_frag(kvs, Ldim, c, kc * 32, hi);
      const v16bf b1 = lds_bt_frag(kvs + 16 * Ldim, Ldim, c, kc * 32, hi);
      s0 = wmma_bf16(aF, b0, s0);
      s1 = wmma_bf16(aF, b1, s1);
    }

    // ---- online causal softmax (half-wave row reductions) ----
    const bool diag = (kb == kbmax);
    float scl[8], p0[8], p1[8];
#pragma unroll
    for (int i = 0; i < 8; ++i) {
      float x0 = s0[i] * sc, x1 = s1[i] * sc;
      if (diag) {
        const int rg = qt * 16 + hi * 8 + i;
        if (kb * 32 + c > rg)      x0 = -1e30f;
        if (kb * 32 + 16 + c > rg) x1 = -1e30f;
      }
      float mx = fmaxf(x0, x1);
      mx = fmaxf(mx, __shfl_xor(mx, 1, 32));
      mx = fmaxf(mx, __shfl_xor(mx, 2, 32));
      mx = fmaxf(mx, __shfl_xor(mx, 4, 32));
      mx = fmaxf(mx, __shfl_xor(mx, 8, 32));
      const float mn = fmaxf(mrow[i], mx);
      scl[i] = __expf(mrow[i] - mn);
      const float e0 = __expf(x0 - mn);
      const float e1 = __expf(x1 - mn);
      float rs = e0 + e1;
      rs += __shfl_xor(rs, 1, 32);
      rs += __shfl_xor(rs, 2, 32);
      rs += __shfl_xor(rs, 4, 32);
      rs += __shfl_xor(rs, 8, 32);
      lrow[i] = lrow[i] * scl[i] + rs;
      mrow[i] = mn;
      p0[i] = e0; p1[i] = e1;
    }
#pragma unroll
    for (int t = 0; t < 18; ++t)
#pragma unroll
      for (int i = 0; i < 8; ++i) acc[t][i] *= scl[i];

    // ---- P: C layout -> A layout via private LDS tile ----
#pragma unroll
    for (int i = 0; i < 8; ++i) {
      const int r = hi * 8 + i;
      ps[r * 40 + c]      = (bf16)p0[i];
      ps[r * 40 + 16 + c] = (bf16)p1[i];
    }
    asm volatile("s_wait_dscnt 0x0" ::: "memory");

    const v16bf aP = lds_a_frag(ps, 40, c, 0, hi);
#pragma unroll
    for (int t = 0; t < 18; ++t) {
      const v16bf bF = lds_b_rowmajor_frag(kvs, Ldim, t * 16 + c, hi);
      acc[t] = wmma_bf16(aP, bF, acc[t]);
    }
  }

  // ---- normalize and store y[B,H,T,L] ----
  float inv[8];
#pragma unroll
  for (int i = 0; i < 8; ++i) inv[i] = 1.f / lrow[i];
  const size_t ybase = ((size_t)(b * Hdim + h) * Tdim + qt * 16) * Ldim;
#pragma unroll
  for (int t = 0; t < 18; ++t) {
    const int l = t * 16 + c;
#pragma unroll
    for (int i = 0; i < 8; ++i) {
      const int r = hi * 8 + i;
      y[ybase + (size_t)r * Ldim + l] = (bf16)(acc[t][i] * inv[i]);
    }
  }
}

// ---------------------------------------------------------------------------
// launch
// ---------------------------------------------------------------------------
static inline size_t align256(size_t x) { return (x + 255) & ~(size_t)255; }

extern "C" void kernel_launch(void* const* d_in, const int* in_sizes, int n_in,
                              void* d_out, int out_size, void* d_ws, size_t ws_size,
                              hipStream_t stream) {
  (void)in_sizes; (void)n_in; (void)out_size; (void)ws_size;
  const float* x        = (const float*)d_in[0];
  const float* latent_w = (const float*)d_in[1];
  const float* latent_b = (const float*)d_in[2];
  const float* Wd_w     = (const float*)d_in[3];
  const float* Wd_b     = (const float*)d_in[4];
  const float* out_w    = (const float*)d_in[5];
  const float* out_b    = (const float*)d_in[6];
  float* out = (float*)d_out;

  char* p = (char*)d_ws;
  const size_t nX  = (size_t)Bdim * Tdim * Cdim;
  const size_t nLW = (size_t)Cdim * Ldim;
  const size_t nWD = (size_t)Hdim * Cdim * Ldim;
  const size_t nOW = (size_t)Hdim * Ldim * Cdim;
  const size_t nKV = (size_t)Bdim * Tdim * Ldim;
  const size_t nQ  = (size_t)Bdim * Hdim * Tdim * Ldim;
  bf16* xb   = (bf16*)p; p += align256(nX  * sizeof(bf16));
  bf16* lwb  = (bf16*)p; p += align256(nLW * sizeof(bf16));
  bf16* wdwb = (bf16*)p; p += align256(nWD * sizeof(bf16));
  bf16* owb  = (bf16*)p; p += align256(nOW * sizeof(bf16));
  bf16* kvb  = (bf16*)p; p += align256(nKV * sizeof(bf16));
  bf16* qb   = (bf16*)p; p += align256(nQ  * sizeof(bf16));
  bf16* yb   = (bf16*)p; p += align256(nQ  * sizeof(bf16));

  auto cvt = [&](const float* src, bf16* dst, size_t n) {
    cvt_f32_bf16<<<dim3((unsigned)((n + 255) / 256)), 256, 0, stream>>>(src, dst, (int)n);
  };
  cvt(x, xb, nX);
  cvt(latent_w, lwb, nLW);
  cvt(Wd_w, wdwb, nWD);
  cvt(out_w, owb, nOW);

  const int Mrow = Bdim * Tdim;  // 4096

  gemm_wmma_bf16<0><<<dim3((Ldim + 127) / 128, Mrow / 64, 1), 256, 0, stream>>>(
      xb, lwb, latent_b, kvb, Mrow, Ldim, Cdim);
  gemm_wmma_bf16<1><<<dim3((Ldim + 127) / 128, Mrow / 64, Hdim), 256, 0, stream>>>(
      xb, wdwb, Wd_b, qb, Mrow, Ldim, Cdim);
  mla_flash_attn<<<dim3(Bdim * Hdim * (Tdim / 16)), 32, 0, stream>>>(qb, kvb, yb);
  gemm_wmma_bf16<2><<<dim3(Cdim / 128, Mrow / 64, 1), 256, 0, stream>>>(
      yb, owb, out_b, out, Mrow, Cdim, Hdim * Ldim);
}
